// DSSMamba_13039520711229
// MI455X (gfx1250) — matrix operand order repaired and checked by reading, not compiled, not run on hardware
//
#include <hip/hip_runtime.h>
#include <hip/hip_bf16.h>

// ---------------------------------------------------------------------------
// Bidirectional Mamba block for MI455X (gfx1250, wave32, WMMA).
//   1) f32->bf16 converts (hidden, W_in, Wx_f/b, W_out)
//   2) WMMA bf16 GEMM (NT=4): xz = hidden @ W_in^T       (4096 x 2048 x 256)
//   3) prep: xbf = bf16(silu(x)), backward flip fused
//   4) WMMA bf16 GEMM (NT=3) per branch: x_dbl = silu(x) @ Wx^T
//   5) chunked selective scan (associative linear recurrence):
//        pass1: per-chunk transition product P and zero-start state S
//        pass2: sequential chunk composition -> H_in per chunk
//        pass3: parallel rescan per chunk from exact H_in, y*silu(z) -> bf16
//      Each block stages its 12KB x_dbl chunk into LDS once via
//      global_load_async_to_lds_b128 (ASYNCcnt) -- the serial loop then reads
//      broadcasts from LDS instead of re-gathering from global per step.
//   6) WMMA bf16 GEMM (NT=4): out = y @ W_out^T          (4096 x 256 x 1024)
// ---------------------------------------------------------------------------

#define D_MODEL  256
#define D_INNER  512
#define D_STATE  16
#define DT_RANK  16
#define NBATCH   2
#define SEQLEN   2048
#define E_DIM    (4 * D_INNER)           // 2048
#define R_DIM    (DT_RANK + 2 * D_STATE) // 48
#define MROWS    (NBATCH * SEQLEN)       // 4096
#define NCHUNK   32
#define CHUNKT   (SEQLEN / NCHUNK)       // 64

typedef __bf16 bf16_t;
typedef __attribute__((ext_vector_type(16))) __bf16 v16bf;
typedef __attribute__((ext_vector_type(8)))  float  v8f;

__device__ __forceinline__ bf16_t f32_to_bf16(float f) {
    union { float f; unsigned int u; } x; x.f = f;
    unsigned int r = x.u + 0x7FFFu + ((x.u >> 16) & 1u); // round-nearest-even
    union { unsigned short s; bf16_t b; } y;
    y.s = (unsigned short)(r >> 16);
    return y.b;
}

__device__ __forceinline__ float siluf(float x) {
    return x / (1.f + __expf(-x));
}

// ---- gfx1250 async global->LDS copy (ASYNCcnt-tracked, no VGPR transit) ----
__device__ __forceinline__ void async_b128_to_lds(void* lds_dst,
                                                  const void* gsrc) {
    // flat shared-aperture address: low 32 bits are the LDS byte offset
    unsigned loff = (unsigned)(size_t)lds_dst;
    asm volatile("global_load_async_to_lds_b128 %0, %1, off"
                 :: "v"(loff), "v"(gsrc) : "memory");
}
__device__ __forceinline__ void wait_async_lds() {
    asm volatile("s_wait_asynccnt 0x0" ::: "memory");
}

// ---------------------------------------------------------------------------
// f32 -> bf16 conversion (grid-stride)
// ---------------------------------------------------------------------------
__global__ void cvt_f32_bf16(const float* __restrict__ src,
                             bf16_t* __restrict__ dst, int n) {
    for (int i = blockIdx.x * blockDim.x + threadIdx.x; i < n;
         i += gridDim.x * blockDim.x)
        dst[i] = f32_to_bf16(src[i]);
}

// ---------------------------------------------------------------------------
// bf16 WMMA GEMM: C[M,N] f32 = A[M,K] (row-major) * B[N,K]^T.
// One wave32 computes a 16 x (16*NT) strip: one A fragment reused for NT
// WMMAs per K-step. Fragment layouts per CDNA5 ISA 7.12.2.
// ---------------------------------------------------------------------------
template <int NT>
__global__ void gemm_bf16_wmma(const bf16_t* __restrict__ A,
                               const bf16_t* __restrict__ B,
                               float* __restrict__ C,
                               int M, int N, int K) {
    const int lane   = threadIdx.x & 31;
    const int wave   = threadIdx.x >> 5;
    const int tilesN = N / (16 * NT);
    const int tile   = blockIdx.x * (blockDim.x >> 5) + wave;
    const int tm     = tile / tilesN;
    const int tg     = tile - tm * tilesN;
    if (tm * 16 >= M) return;

    const int half = lane >> 4;   // 0: lanes 0-15, 1: lanes 16-31
    const int lrow = lane & 15;

    const unsigned int* __restrict__ Arow =
        reinterpret_cast<const unsigned int*>(A + (size_t)(tm * 16 + lrow) * K);
    const unsigned int* Brow[NT];
    #pragma unroll
    for (int nt = 0; nt < NT; ++nt)
        Brow[nt] = reinterpret_cast<const unsigned int*>(
            B + (size_t)(tg * 16 * NT + nt * 16 + lrow) * K);

    v8f acc[NT];
    #pragma unroll
    for (int nt = 0; nt < NT; ++nt) acc[nt] = (v8f){};

    for (int k0 = 0; k0 < K; k0 += 32) {
        const int kw = k0 >> 1;                   // index in bf16-pairs
        __builtin_prefetch(Arow + kw + 32, 0, 1); // global_prefetch_b8

        union { v16bf v; unsigned int u[8]; } a;
        #pragma unroll
        for (int j = 0; j < 4; ++j) {
            a.u[j]     = Arow[kw + half * 4 + j];     // K = k0 + half*8 + 2j
            a.u[4 + j] = Arow[kw + 8 + half * 4 + j]; // K = k0+16 + half*8 + 2j
        }
        #pragma unroll
        for (int nt = 0; nt < NT; ++nt) {
            union { v16bf v; unsigned int u[8]; } b;
            #pragma unroll
            for (int j = 0; j < 4; ++j) {
                b.u[j]     = Brow[nt][kw + half * 4 + j];     // K = k0 + half*8 + 2j
                b.u[4 + j] = Brow[nt][kw + 8 + half * 4 + j]; // K = k0+16 + half*8 + 2j
            }
            acc[nt] = __builtin_amdgcn_wmma_f32_16x16x32_bf16(
                false, a.v, false, b.v, (short)0, acc[nt], false, false);
        }
    }

    #pragma unroll
    for (int nt = 0; nt < NT; ++nt) {
        float* __restrict__ Crow =
            C + (size_t)(tm * 16 + half * 8) * N + tg * 16 * NT + nt * 16 + lrow;
        #pragma unroll
        for (int r = 0; r < 8; ++r)
            Crow[(size_t)r * N] = acc[nt][r];
    }
}

// ---------------------------------------------------------------------------
// prep: xbf[branch][b][l][d] = bf16( silu( xz[b][l_eff][branch*1024 + d] ) )
// ---------------------------------------------------------------------------
__global__ void prep_silu_x(const float* __restrict__ xz,
                            bf16_t* __restrict__ xbf) {
    const int i      = blockIdx.x * blockDim.x + threadIdx.x; // 2*2*2048*512
    const int d      = i & (D_INNER - 1);
    const int l      = (i >> 9) & (SEQLEN - 1);
    const int b      = (i >> 20) & 1;
    const int branch = i >> 21;
    const int leff   = branch ? (SEQLEN - 1 - l) : l;
    const int eoff   = branch << 10;
    const float x = xz[((size_t)b * SEQLEN + leff) * E_DIM + eoff + d];
    xbf[i] = f32_to_bf16(siluf(x));
}

// ---------------------------------------------------------------------------
// Chunked selective scan.
// Channel ch = (branch,b,d); chunk c covers l in [c*64, c*64+64).
// P/S/Hin layout: [branch][b][c][n][d]  (n-slices are coalesced 128B lines)
// Every block shares one (branch,b,c): its 64x48 f32 x_dbl chunk (12KB) is
// staged once into LDS with async b128 loads, then broadcast-read per step.
// ---------------------------------------------------------------------------
__device__ __forceinline__ size_t ps_base(int branch, int b, int c, int d) {
    return ((size_t)(((branch * NBATCH + b) * NCHUNK + c) * D_STATE) << 9) + d;
}

__device__ __forceinline__ void stage_xdbl_chunk(float* sxd,
                                                 const float* __restrict__ src,
                                                 int tid) {
    // 64*48 floats = 12288 bytes = 768 x b128; 3 per thread (256 threads)
    const char* s = (const char*)src;
    char*       dch = (char*)sxd;
    #pragma unroll
    for (int k = 0; k < 3; ++k)
        async_b128_to_lds(dch + tid * 16 + k * 4096, s + tid * 16 + k * 4096);
    wait_async_lds();
    __syncthreads();
}

// pass 1: per-chunk transition product P = prod(dA), zero-start state S
__global__ void scan_pass1(const float* __restrict__ xz,    // [B][L][2048]
                           const float* __restrict__ xdbl,  // [2][B][L][48]
                           const float* __restrict__ Wdt_f,
                           const float* __restrict__ Wdt_b,
                           const float* __restrict__ bdt_f,
                           const float* __restrict__ bdt_b,
                           const float* __restrict__ Alog_f,
                           const float* __restrict__ Alog_b,
                           float* __restrict__ P,
                           float* __restrict__ S) {
    __shared__ float sxd[CHUNKT * R_DIM];                     // 12 KB
    const int idx    = blockIdx.x * blockDim.x + threadIdx.x; // 65536
    const int d      = idx & (D_INNER - 1);
    const int c      = (idx >> 9) & (NCHUNK - 1);
    const int b      = (idx >> 14) & 1;
    const int branch = idx >> 15;
    const int l0     = c * CHUNKT;

    const float* __restrict__ xdbase =
        xdbl + (size_t)(branch * NBATCH + b) * SEQLEN * R_DIM;
    stage_xdbl_chunk(sxd, xdbase + (size_t)l0 * R_DIM, threadIdx.x);

    const float* __restrict__ Wdt  = branch ? Wdt_b  : Wdt_f;
    const float* __restrict__ bdt  = branch ? bdt_b  : bdt_f;
    const float* __restrict__ Alog = branch ? Alog_b : Alog_f;

    float Av[D_STATE], wdt[DT_RANK], h[D_STATE], p[D_STATE];
    #pragma unroll
    for (int n = 0; n < D_STATE; ++n) {
        Av[n]  = -__expf(Alog[d * D_STATE + n]);
        wdt[n] = Wdt[d * DT_RANK + n];
        h[n]   = 0.f;
        p[n]   = 1.f;
    }
    const float bias = bdt[d];
    const int   eoff = branch << 10;
    const float* __restrict__ xzb = xz + (size_t)b * SEQLEN * E_DIM;

    for (int t = 0; t < CHUNKT; ++t) {
        const int l    = l0 + t;
        const int leff = branch ? (SEQLEN - 1 - l) : l;
        const float* xd = &sxd[t * R_DIM];

        const float u = siluf(xzb[(size_t)leff * E_DIM + eoff + d]);
        float tt = bias;
        #pragma unroll
        for (int r = 0; r < DT_RANK; ++r) tt = fmaf(wdt[r], xd[r], tt);
        const float dt = (tt > 20.f) ? tt : log1pf(__expf(tt));

        #pragma unroll
        for (int n = 0; n < D_STATE; ++n) {
            const float dA = __expf(dt * Av[n]);
            h[n] = fmaf(dA, h[n], dt * xd[DT_RANK + n] * u);
            p[n] *= dA;
        }
    }
    const size_t base = ps_base(branch, b, c, d);
    #pragma unroll
    for (int n = 0; n < D_STATE; ++n) {
        P[base + ((size_t)n << 9)] = p[n];
        S[base + ((size_t)n << 9)] = h[n];
    }
}

// pass 2: sequential chunk composition H_in(c+1) = P_c*H_in(c) + S_c
__global__ void scan_pass2(const float* __restrict__ P,
                           const float* __restrict__ S,
                           float* __restrict__ Hin) {
    const int ch     = blockIdx.x * blockDim.x + threadIdx.x; // 2048
    const int d      = ch & (D_INNER - 1);
    const int b      = (ch >> 9) & 1;
    const int branch = ch >> 10;

    float h[D_STATE];
    #pragma unroll
    for (int n = 0; n < D_STATE; ++n) h[n] = 0.f;

    for (int c = 0; c < NCHUNK; ++c) {
        const size_t base = ps_base(branch, b, c, d);
        #pragma unroll
        for (int n = 0; n < D_STATE; ++n) {
            const size_t o = base + ((size_t)n << 9);
            Hin[o] = h[n];
            h[n] = fmaf(P[o], h[n], S[o]);
        }
    }
}

// pass 3: rescan each chunk from exact H_in, emit y*silu(z) as bf16
__global__ void scan_pass3(const float* __restrict__ xz,
                           const float* __restrict__ xdbl,
                           const float* __restrict__ Wdt_f,
                           const float* __restrict__ Wdt_b,
                           const float* __restrict__ bdt_f,
                           const float* __restrict__ bdt_b,
                           const float* __restrict__ Alog_f,
                           const float* __restrict__ Alog_b,
                           const float* __restrict__ Df,
                           const float* __restrict__ Db,
                           const float* __restrict__ Hin,
                           bf16_t* __restrict__ ybf) {       // [B][L][1024]
    __shared__ float sxd[CHUNKT * R_DIM];                     // 12 KB
    const int idx    = blockIdx.x * blockDim.x + threadIdx.x; // 65536
    const int d      = idx & (D_INNER - 1);
    const int c      = (idx >> 9) & (NCHUNK - 1);
    const int b      = (idx >> 14) & 1;
    const int branch = idx >> 15;
    const int l0     = c * CHUNKT;

    const float* __restrict__ xdbase =
        xdbl + (size_t)(branch * NBATCH + b) * SEQLEN * R_DIM;
    stage_xdbl_chunk(sxd, xdbase + (size_t)l0 * R_DIM, threadIdx.x);

    const float* __restrict__ Wdt  = branch ? Wdt_b  : Wdt_f;
    const float* __restrict__ bdt  = branch ? bdt_b  : bdt_f;
    const float* __restrict__ Alog = branch ? Alog_b : Alog_f;
    const float* __restrict__ Dp   = branch ? Db     : Df;

    float Av[D_STATE], wdt[DT_RANK], h[D_STATE];
    const size_t base = ps_base(branch, b, c, d);
    #pragma unroll
    for (int n = 0; n < D_STATE; ++n) {
        Av[n]  = -__expf(Alog[d * D_STATE + n]);
        wdt[n] = Wdt[d * DT_RANK + n];
        h[n]   = Hin[base + ((size_t)n << 9)];
    }
    const float bias = bdt[d];
    const float Dd   = Dp[d];
    const int   eoff = branch << 10;
    const float* __restrict__ xzb = xz + (size_t)b * SEQLEN * E_DIM;
    bf16_t* __restrict__ yb =
        ybf + (size_t)b * SEQLEN * (2 * D_INNER) + (branch << 9) + d;

    for (int t = 0; t < CHUNKT; ++t) {
        const int l    = l0 + t;
        const int leff = branch ? (SEQLEN - 1 - l) : l;
        const float* xd = &sxd[t * R_DIM];

        const float u = siluf(xzb[(size_t)leff * E_DIM + eoff + d]);
        float tt = bias;
        #pragma unroll
        for (int r = 0; r < DT_RANK; ++r) tt = fmaf(wdt[r], xd[r], tt);
        const float dt = (tt > 20.f) ? tt : log1pf(__expf(tt));

        float y = 0.f;
        #pragma unroll
        for (int n = 0; n < D_STATE; ++n) {
            const float dA = __expf(dt * Av[n]);
            h[n] = fmaf(dA, h[n], dt * xd[DT_RANK + n] * u);
            y    = fmaf(h[n], xd[DT_RANK + D_STATE + n], y);
        }
        y = fmaf(u, Dd, y);

        const float zr = xzb[(size_t)leff * E_DIM + eoff + D_INNER + d];
        y *= siluf(zr);
        yb[(size_t)leff * (2 * D_INNER)] = f32_to_bf16(y);
    }
}

// ---------------------------------------------------------------------------
// Host-side launch
// ---------------------------------------------------------------------------
extern "C" void kernel_launch(void* const* d_in, const int* in_sizes, int n_in,
                              void* d_out, int out_size, void* d_ws, size_t ws_size,
                              hipStream_t stream) {
    (void)in_sizes; (void)n_in; (void)out_size; (void)ws_size;

    const float* hidden = (const float*)d_in[0];
    const float* W_in   = (const float*)d_in[1];
    const float* Wx_f   = (const float*)d_in[2];
    const float* Wx_b   = (const float*)d_in[3];
    const float* Wdt_f  = (const float*)d_in[4];
    const float* Wdt_b  = (const float*)d_in[5];
    const float* bdt_f  = (const float*)d_in[6];
    const float* bdt_b  = (const float*)d_in[7];
    const float* Alog_f = (const float*)d_in[8];
    const float* Alog_b = (const float*)d_in[9];
    const float* D_f    = (const float*)d_in[10];
    const float* D_b    = (const float*)d_in[11];
    const float* W_out  = (const float*)d_in[12];
    float* out = (float*)d_out;                    // [2,2048,256]

    // ---- workspace carve-up ----
    char* ws = (char*)d_ws;
    size_t off = 0;
    bf16_t* hbf    = (bf16_t*)(ws + off); off += (size_t)MROWS * D_MODEL * 2;
    bf16_t* winbf  = (bf16_t*)(ws + off); off += (size_t)E_DIM * D_MODEL * 2;
    bf16_t* wxbf   = (bf16_t*)(ws + off); off += (size_t)2 * R_DIM * D_INNER * 2;
    bf16_t* woutbf = (bf16_t*)(ws + off); off += (size_t)D_MODEL * 2 * D_INNER * 2;
    float*  xz     = (float*) (ws + off); off += (size_t)MROWS * E_DIM * 4;
    bf16_t* xbf    = (bf16_t*)(ws + off); off += (size_t)2 * MROWS * D_INNER * 2;
    float*  xdbl   = (float*) (ws + off); off += (size_t)2 * MROWS * R_DIM * 4;
    bf16_t* ybf    = (bf16_t*)(ws + off); off += (size_t)MROWS * 2 * D_INNER * 2;
    const size_t psN = (size_t)2 * NBATCH * NCHUNK * D_STATE * D_INNER;
    float* Pbuf = (float*)(ws + off); off += psN * 4;
    float* Sbuf = (float*)(ws + off); off += psN * 4;
    float* Hin  = (float*)(ws + off); off += psN * 4;

    // ---- 1) bf16 converts ----
    {
        int n;
        n = MROWS * D_MODEL;
        cvt_f32_bf16<<<(n + 255) / 256, 256, 0, stream>>>(hidden, hbf, n);
        n = E_DIM * D_MODEL;
        cvt_f32_bf16<<<(n + 255) / 256, 256, 0, stream>>>(W_in, winbf, n);
        n = R_DIM * D_INNER;
        cvt_f32_bf16<<<(n + 255) / 256, 256, 0, stream>>>(Wx_f, wxbf, n);
        cvt_f32_bf16<<<(n + 255) / 256, 256, 0, stream>>>(Wx_b, wxbf + n, n);
        n = D_MODEL * 2 * D_INNER;
        cvt_f32_bf16<<<(n + 255) / 256, 256, 0, stream>>>(W_out, woutbf, n);
    }

    // ---- 2) input projection: xz = hidden @ W_in^T ----
    {
        const int M = MROWS, N = E_DIM, K = D_MODEL;
        const int tiles = (M / 16) * (N / 64);          // 8192
        gemm_bf16_wmma<4><<<tiles / 4, 128, 0, stream>>>(hbf, winbf, xz, M, N, K);
    }

    // ---- 3) silu + flip into bf16 ----
    {
        const int n = 2 * MROWS * D_INNER;
        prep_silu_x<<<n / 256, 256, 0, stream>>>(xz, xbf);
    }

    // ---- 4) x_dbl GEMMs per branch ----
    {
        const int M = MROWS, N = R_DIM, K = D_INNER;
        const int tiles = (M / 16) * (N / 48);          // 256
        gemm_bf16_wmma<3><<<tiles / 4, 128, 0, stream>>>(xbf, wxbf, xdbl, M, N, K);
        gemm_bf16_wmma<3><<<tiles / 4, 128, 0, stream>>>(
            xbf + (size_t)MROWS * D_INNER, wxbf + (size_t)R_DIM * D_INNER,
            xdbl + (size_t)MROWS * R_DIM, M, N, K);
    }

    // ---- 5) chunked selective scan ----
    scan_pass1<<<(2 * NBATCH * NCHUNK * D_INNER) / 256, 256, 0, stream>>>(
        xz, xdbl, Wdt_f, Wdt_b, bdt_f, bdt_b, Alog_f, Alog_b, Pbuf, Sbuf);
    scan_pass2<<<(2 * NBATCH * D_INNER) / 256, 256, 0, stream>>>(Pbuf, Sbuf, Hin);
    scan_pass3<<<(2 * NBATCH * NCHUNK * D_INNER) / 256, 256, 0, stream>>>(
        xz, xdbl, Wdt_f, Wdt_b, bdt_f, bdt_b, Alog_f, Alog_b, D_f, D_b, Hin, ybf);

    // ---- 6) output projection: out = y @ W_out^T ----
    {
        const int M = MROWS, N = D_MODEL, K = 2 * D_INNER;
        const int tiles = (M / 16) * (N / 64);          // 1024
        gemm_bf16_wmma<4><<<tiles / 4, 128, 0, stream>>>(ybf, woutbf, out, M, N, K);
    }
}
